// Net_39951785787542
// MI455X (gfx1250) — matrix operand order repaired and checked by reading, not compiled, not run on hardware
//
#include <hip/hip_runtime.h>
#include <hip/hip_bf16.h>

// ---------------- problem constants ----------------
#define Nn  200000
#define En  400000
#define Pn  800000
#define Cn  50000
#define C1n 300000
#define P2n 600000
#define Bn  4096
#define Dn  128

// ---------------- param offsets (flattened JAX pytree, dicts sorted by key) ----
#define OFF_ATOM    0
#define OFF_CYCEMB  2816
#define OFF_CYCEPS  5632
#define OFF_EDGEEMB 5633
#define OFF_LAYERS  6145
#define LAYER_SZ    413445
#define OFF_LINW    1659925
#define OFF_LINB    1660181
#define OFF_POOL    1660182
#define POOL_SZ     33280
// per-layer (sorted: ec, mlp, ne)
#define EC_AFF      0
#define EC_EPS1     1
#define EC_EPS2     2
#define EC_LIFT_W1  3
#define EC_LIFT_W2  65539
#define EC_LIFT_B1  98307
#define EC_LIFT_B2  98563
#define EC_LIFT_G1  98691
#define EC_LIFT_G2  98947
#define EC_LVL1_W   99075
#define EC_LVL1_B   148227
#define EC_LVL1_G   148355
#define EC_LVL2_W1  148483
#define EC_LVL2_W2  181251
#define EC_LVL2_B1  214019
#define EC_LVL2_B2  214275
#define EC_LVL2_G1  214403
#define EC_LVL2_G2  214659
#define MLP_W       214787
#define MLP_B       247555
#define MLP_G       247683
#define NE_EPS1     247811
#define NE_EPS2     247812
#define NE_LIFT_W1  247813
#define NE_LIFT_W2  280581
#define NE_LIFT_B1  313349
#define NE_LIFT_B2  313605
#define NE_LIFT_G1  313733
#define NE_LIFT_G2  313989
#define NE_LVL1_W   314117
#define NE_LVL1_B   346885
#define NE_LVL1_G   347013
#define NE_LVL2_W1  347141
#define NE_LVL2_W2  379909
#define NE_LVL2_B1  412677
#define NE_LVL2_B2  412933
#define NE_LVL2_G1  413061
#define NE_LVL2_G2  413317

typedef __attribute__((ext_vector_type(16))) __bf16 bf16x16;
typedef __attribute__((ext_vector_type(8)))  float  floatx8;

static __device__ __forceinline__ unsigned short f2bf(float x) {
    unsigned u = __builtin_bit_cast(unsigned, x);
    unsigned r = u + 0x7FFFu + ((u >> 16) & 1u);   // round-to-nearest-even
    return (unsigned short)(r >> 16);
}

union AFrag { unsigned u[8]; bf16x16 v; };

// ================= WMMA GEMM: Y[M,Nc] = concat_K(X0|X1|X2) @ W, + column stats ==
// Two-stage software pipeline: stage K-chunk kc+1 into the alternate LDS buffer
// while the WMMAs for chunk kc consume the current one (one barrier/iteration).
__global__ __launch_bounds__(256)
void k_gemm(const float* __restrict__ X0, int ld0,
            const float* __restrict__ X1, int ld1,
            const float* __restrict__ X2, int ld2,
            int nseg, const float* __restrict__ W, int Nc,
            float* __restrict__ Y, float* __restrict__ colsum,
            float* __restrict__ colsq, int M) {
    __shared__ __align__(16) unsigned short Ash[2][128 * 34];
    __shared__ __align__(16) unsigned short Bsh[2][64 * 34];
    __shared__ float redS[8 * 64];
    __shared__ float redQ[8 * 64];

    const int t    = threadIdx.x;
    const int wave = t >> 5;
    const int lane = t & 31;
    const int half = lane >> 4;
    const int lr   = lane & 15;
    const int colb = blockIdx.y * 64;

    // ---- hoisted staging addresses (loop-invariant per segment) ----
    const int ar = t >> 1;                 // A: row 0..127
    const int ac = (t & 1) << 4;           // A: k sub-offset 0/16
    const size_t arow = (size_t)blockIdx.x * 128 + ar;
    const bool arow_ok = arow < (size_t)M;
    const float* abase[3];
    abase[0] = X0 + arow * (size_t)ld0 + ac;
    abase[1] = (nseg > 1) ? (X1 + arow * (size_t)ld1 + ac) : abase[0];
    abase[2] = (nseg > 2) ? (X2 + arow * (size_t)ld2 + ac) : abase[0];

    const int bk = t >> 3;                 // B: k row 0..31
    const int bn = (t & 7) << 3;           // B: col 0..56
    const float* wbase = W + (size_t)bk * Nc + colb + bn;

    auto stage = [&](int buf, int kc) {
        const int seg = kc >> 2;
        const int kl  = (kc & 3) << 5;
        unsigned short* adst = &Ash[buf][ar * 34 + ac];
        if (arow_ok) {
            const float* p = abase[seg] + kl;
#pragma unroll
            for (int j = 0; j < 16; j += 4) {
                float4 v = *(const float4*)(p + j);
                adst[j + 0] = f2bf(v.x); adst[j + 1] = f2bf(v.y);
                adst[j + 2] = f2bf(v.z); adst[j + 3] = f2bf(v.w);
            }
        } else {
#pragma unroll
            for (int j = 0; j < 16; ++j) adst[j] = 0;
        }
        const float* wp = wbase + (size_t)kc * 32 * Nc;
        unsigned short* bdst = &Bsh[buf][0];
#pragma unroll
        for (int j = 0; j < 8; j += 4) {
            float4 v = *(const float4*)(wp + j);
            bdst[(bn + j + 0) * 34 + bk] = f2bf(v.x);
            bdst[(bn + j + 1) * 34 + bk] = f2bf(v.y);
            bdst[(bn + j + 2) * 34 + bk] = f2bf(v.z);
            bdst[(bn + j + 3) * 34 + bk] = f2bf(v.w);
        }
    };

    floatx8 acc[4] = {};
    const int kcCount = nseg * 4;

    stage(0, 0);
    __syncthreads();
    for (int kc = 0; kc < kcCount; ++kc) {
        const int buf = kc & 1;
        if (kc + 1 < kcCount) stage(buf ^ 1, kc + 1);
        // --- A fragment (16x32 bf16, CDNA5 layout) ---
        AFrag a;
        {
            const unsigned short* arw = &Ash[buf][(wave * 16 + lr) * 34 + half * 8];
#pragma unroll
            for (int v = 0; v < 4; ++v) a.u[v] = *(const unsigned*)(arw + 2 * v);
#pragma unroll
            for (int v = 0; v < 4; ++v) a.u[4 + v] = *(const unsigned*)(arw + 16 + 2 * v);
        }
#pragma unroll
        for (int nt = 0; nt < 4; ++nt) {
            AFrag b;
            const unsigned short* bc = &Bsh[buf][(nt * 16 + lr) * 34 + half * 8];
#pragma unroll
            for (int v = 0; v < 8; ++v) b.u[v] = *(const unsigned*)(bc + (v >> 2) * 16 + 2 * (v & 3));
            acc[nt] = __builtin_amdgcn_wmma_f32_16x16x32_bf16(
                false, a.v, false, b.v, (short)0, acc[nt], false, false);
        }
        __syncthreads();
    }

    // --- epilogue: store + per-column partial stats ---
    const size_t rbase = (size_t)blockIdx.x * 128 + wave * 16 + half * 8;
#pragma unroll
    for (int nt = 0; nt < 4; ++nt) {
        const int col = colb + nt * 16 + lr;
        float ps = 0.f, pq = 0.f;
#pragma unroll
        for (int rr = 0; rr < 8; ++rr) {
            float v = acc[nt][rr];
            size_t row = rbase + rr;
            if (row < (size_t)M) Y[row * (size_t)Nc + col] = v;
            ps += v; pq += v * v;
        }
        ps += __shfl_xor(ps, 16, 32);
        pq += __shfl_xor(pq, 16, 32);
        if (half == 0) {
            redS[wave * 64 + nt * 16 + lr] = ps;
            redQ[wave * 64 + nt * 16 + lr] = pq;
        }
    }
    __syncthreads();
    if (t < 64) {
        float s = 0.f, q = 0.f;
#pragma unroll
        for (int w = 0; w < 8; ++w) { s += redS[w * 64 + t]; q += redQ[w * 64 + t]; }
        atomicAdd(&colsum[colb + t], s);
        atomicAdd(&colsq[colb + t], q);
    }
}

// ================= BatchNorm(train-stats) + ReLU (optionally accumulate) =======
__global__ void k_bn_relu(float* __restrict__ out, const float* __restrict__ Y,
                          const float* __restrict__ s, const float* __restrict__ q,
                          const float* __restrict__ g, const float* __restrict__ b,
                          int M, int Nc, int accf, long long n) {
    size_t i = (size_t)blockIdx.x * blockDim.x + threadIdx.x;
    if (i >= (size_t)n) return;
    int col = (int)(i % (size_t)Nc);
    float invM = 1.0f / (float)M;
    float mean = s[col] * invM;
    float var  = q[col] * invM - mean * mean;
    float y = (Y[i] - mean) * rsqrtf(var + 1e-5f) * g[col] + b[col];
    y = fmaxf(y, 0.0f);
    if (accf) out[i] += y; else out[i] = y;
}

// ================= scatter-add rows: dst[di[p]] += src[si[p] or p] ============
__global__ void k_scatter(float* __restrict__ dst, const float* __restrict__ src,
                          const int* __restrict__ si, const int* __restrict__ di,
                          int np) {
    size_t t = (size_t)blockIdx.x * blockDim.x + threadIdx.x;
    int p = (int)(t >> 5);
    int lane = (int)(t & 31);
    if (p >= np) return;
    int srow = si ? si[p] : p;
    const float* s = src + (size_t)srow * Dn + lane * 4;
    float* d = dst + (size_t)di[p] * Dn + lane * 4;
    float4 v = *(const float4*)s;
    atomicAdd(d + 0, v.x); atomicAdd(d + 1, v.y);
    atomicAdd(d + 2, v.z); atomicAdd(d + 3, v.w);
}

__global__ void k_embed(float* __restrict__ out, const float* __restrict__ emb,
                        const int* __restrict__ idx, long long n) {
    size_t i = (size_t)blockIdx.x * blockDim.x + threadIdx.x;
    if (i >= (size_t)n) return;
    out[i] = emb[(size_t)idx[i >> 7] * Dn + (i & 127)];
}

__global__ void k_embed2(float* __restrict__ out, const float* __restrict__ emb,
                         const int* __restrict__ idx1, const int* __restrict__ idx2,
                         long long n) {
    size_t i = (size_t)blockIdx.x * blockDim.x + threadIdx.x;
    if (i >= (size_t)n) return;
    out[i] = emb[(size_t)idx2[idx1[i >> 7]] * Dn + (i & 127)];
}

__global__ void k_gather(float* __restrict__ out, const float* __restrict__ src,
                         const int* __restrict__ idx, long long n) {
    size_t i = (size_t)blockIdx.x * blockDim.x + threadIdx.x;
    if (i >= (size_t)n) return;
    out[i] = src[(size_t)idx[i >> 7] * Dn + (i & 127)];
}

__global__ void k_cyc_init(float* __restrict__ cyc, const float* __restrict__ xe,
                           const float* __restrict__ segC, const int* __restrict__ dom,
                           const float* __restrict__ ceP, long long n) {
    size_t i = (size_t)blockIdx.x * blockDim.x + threadIdx.x;
    if (i >= (size_t)n) return;
    float ce = *ceP;
    cyc[i] = ce * xe[i] + (1.0f - ce) * segC[(size_t)dom[i >> 7] * Dn + (i & 127)];
}

__global__ void k_comb2(float* __restrict__ out, const float* __restrict__ x,
                        const float* __restrict__ y, const float* __restrict__ epsP,
                        long long n) {
    size_t i = (size_t)blockIdx.x * blockDim.x + threadIdx.x;
    if (i >= (size_t)n) return;
    out[i] = (1.0f + *epsP) * x[i] + y[i];
}

__global__ void k_comb3(float* __restrict__ out, const float* __restrict__ e,
                        const float* __restrict__ g0, const float* __restrict__ g1,
                        const float* __restrict__ epsP, const float* __restrict__ aeP,
                        long long n) {
    size_t i = (size_t)blockIdx.x * blockDim.x + threadIdx.x;
    if (i >= (size_t)n) return;
    float ae = *aeP;
    out[i] = (1.0f + *epsP) * e[i] + ae * g0[i] + (1.0f - ae) * g1[i];
}

__global__ void k_pre2(float* __restrict__ aB, float* __restrict__ dB,
                       const float* __restrict__ cyc, const float* __restrict__ segC,
                       const float* __restrict__ cnt, const int* __restrict__ dom,
                       const float* __restrict__ epsP, long long n) {
    size_t i = (size_t)blockIdx.x * blockDim.x + threadIdx.x;
    if (i >= (size_t)n) return;
    float e = 1.0f + *epsP;
    int dm = dom[i >> 7];
    float m = segC[(size_t)dm * Dn + (i & 127)] / fmaxf(cnt[dm], 1.0f);
    aB[i] = e * cyc[i] + aB[i];
    dB[i] = e * m + dB[i];
}

__global__ void k_count(float* __restrict__ cnt, const int* __restrict__ dom, int n) {
    int i = blockIdx.x * blockDim.x + threadIdx.x;
    if (i >= n) return;
    atomicAdd(&cnt[dom[i]], 1.0f);
}

__global__ void k_final(float* __restrict__ out, const float* __restrict__ rep,
                        const float* __restrict__ w, const float* __restrict__ b0,
                        int nB) {
    int i = blockIdx.x * blockDim.x + threadIdx.x;
    if (i >= nB) return;
    const float* r = rep + (size_t)i * 256;
    float s = 0.f;
#pragma unroll 8
    for (int k = 0; k < 256; ++k) s += r[k] * w[k];
    out[i] = s + b0[0];
}

// =============================== host ==========================================
extern "C" void kernel_launch(void* const* d_in, const int* in_sizes, int n_in,
                              void* d_out, int out_size, void* d_ws, size_t ws_size,
                              hipStream_t stream) {
    const float* prm  = (const float*)d_in[0];
    const int* x      = (const int*)d_in[1];
    const int* eattr  = (const int*)d_in[2];
    const int* n2en   = (const int*)d_in[3];
    const int* n2ee   = (const int*)d_in[4];
    const int* catoms = (const int*)d_in[5];
    const int* cdom   = (const int*)d_in[6];
    const int* e2ce   = (const int*)d_in[7];
    const int* e2cn   = (const int*)d_in[8];
    const int* batch  = (const int*)d_in[9];
    const int* ebatch = (const int*)d_in[10];
    const int* cbatch = (const int*)d_in[11];
    float* out = (float*)d_out;

    // ---- workspace arena ----
    float* Wb = (float*)d_ws;
    size_t o = 0;
    auto alloc = [&](size_t n) { float* p = Wb + o; o += (n + 63) & ~(size_t)63; return p; };
    float* node0 = alloc((size_t)Nn * Dn);
    float* node1 = alloc((size_t)Nn * Dn);
    float* edge0 = alloc((size_t)En * Dn);
    float* edge1 = alloc((size_t)En * Dn);
    float* cyc0  = alloc((size_t)C1n * Dn);
    float* cyc1  = alloc((size_t)C1n * Dn);
    float* bufE1 = alloc((size_t)En * Dn);
    float* bufE2 = alloc((size_t)En * Dn);
    float* bufE3 = alloc((size_t)En * Dn);
    float* bufN  = alloc((size_t)Nn * Dn);
    float* c1a   = alloc((size_t)C1n * Dn);
    float* c1b   = alloc((size_t)C1n * Dn);
    float* c1c   = alloc((size_t)C1n * Dn);
    float* c1d   = alloc((size_t)C1n * Dn);
    float* segC  = alloc((size_t)Cn * Dn);
    float* cnt   = alloc((size_t)Cn);
    float* Ybig  = alloc((size_t)En * 256);
    float* Hbig  = alloc((size_t)En * 256);
    float* colsum = alloc(256);
    float* colsq  = alloc(256);
    float* r0 = alloc((size_t)Bn * Dn);
    float* r1 = alloc((size_t)Bn * Dn);
    float* r2 = alloc((size_t)Bn * Dn);
    float* rep = alloc((size_t)Bn * 256);

    auto MS = [&](float* p, size_t n) { hipMemsetAsync(p, 0, n * sizeof(float), stream); };
    auto NB = [](long long n) { return (unsigned)((n + 255) / 256); };

    auto GEMM = [&](const float* X0, int ld0, const float* X1, int ld1,
                    const float* X2, int ld2, int nseg, const float* Wp, int Nc,
                    float* Yp, int M) {
        MS(colsum, 256); MS(colsq, 256);
        dim3 g((M + 127) / 128, Nc / 64);
        k_gemm<<<g, 256, 0, stream>>>(X0, ld0, X1, ld1, X2, ld2, nseg, Wp, Nc,
                                      Yp, colsum, colsq, M);
    };
    auto BN = [&](float* o_, const float* Yp, const float* g, const float* b,
                  int M, int Nc, int accf) {
        long long n = (long long)M * Nc;
        k_bn_relu<<<NB(n), 256, 0, stream>>>(o_, Yp, colsum, colsq, g, b, M, Nc, accf, n);
    };
    auto SCAT = [&](float* dst, const float* src, const int* si, const int* di, int np) {
        long long n = (long long)np * 32;
        k_scatter<<<NB(n), 256, 0, stream>>>(dst, src, si, di, np);
    };
    // mlp2: Lin(K=nseg*128 -> 2D) -> BN -> ReLU -> Lin(2D -> D) -> BN -> ReLU
    auto MLP2 = [&](const float* X0, int ld0, const float* X1, int ld1, int nseg,
                    const float* base, size_t w1, size_t g1, size_t b1,
                    size_t w2, size_t g2, size_t b2, float* o_, int M) {
        GEMM(X0, ld0, X1, ld1, nullptr, 0, nseg, base + w1, 256, Ybig, M);
        BN(Hbig, Ybig, base + g1, base + b1, M, 256, 0);
        GEMM(Hbig, 256, Hbig + 128, 256, nullptr, 0, 2, base + w2, 128, Ybig, M);
        BN(o_, Ybig, base + g2, base + b2, M, 128, 0);
    };

    // ---- embeddings ----
    k_embed<<<NB((long long)Nn * Dn), 256, 0, stream>>>(node0, prm + OFF_ATOM, x, (long long)Nn * Dn);
    k_embed<<<NB((long long)En * Dn), 256, 0, stream>>>(edge0, prm + OFF_EDGEEMB, eattr, (long long)En * Dn);
    k_embed2<<<NB((long long)C1n * Dn), 256, 0, stream>>>(c1a, prm + OFF_CYCEMB, catoms, x, (long long)C1n * Dn);
    MS(segC, (size_t)Cn * Dn);
    SCAT(segC, c1a, nullptr, cdom, C1n);
    k_cyc_init<<<NB((long long)C1n * Dn), 256, 0, stream>>>(cyc0, c1a, segC, cdom,
                                                            prm + OFF_CYCEPS, (long long)C1n * Dn);

    float *node = node0, *nodeN = node1, *edge = edge0, *edgeN = edge1,
          *cyc = cyc0, *cycN = cyc1;

    for (int l = 0; l < 4; ++l) {
        const float* Lp = prm + OFF_LAYERS + (size_t)l * LAYER_SZ;
        // ---- SplitLayer node<->edge ----
        MS(bufE1, (size_t)En * Dn);
        SCAT(bufE1, node, n2en, n2ee, Pn);                               // lift
        GEMM(bufE1, 128, edge, 128, nullptr, 0, 2, Lp + NE_LVL1_W, 128, Ybig, En);
        BN(bufE2, Ybig, Lp + NE_LVL1_G, Lp + NE_LVL1_B, En, 128, 0);     // lvl
        MS(bufN, (size_t)Nn * Dn);
        SCAT(bufN, bufE2, n2ee, n2en, Pn);                               // lvl_aggr
        k_comb2<<<NB((long long)Nn * Dn), 256, 0, stream>>>(bufN, node, bufN,
                Lp + NE_EPS1, (long long)Nn * Dn);                       // t1
        MLP2(bufN, 128, nullptr, 0, 1, Lp, NE_LVL2_W1, NE_LVL2_G1, NE_LVL2_B1,
             NE_LVL2_W2, NE_LVL2_G2, NE_LVL2_B2, nodeN, Nn);             // node_new
        k_comb2<<<NB((long long)En * Dn), 256, 0, stream>>>(bufE1, edge, bufE1,
                Lp + NE_EPS2, (long long)En * Dn);                       // t2
        MLP2(bufE1, 128, nullptr, 0, 1, Lp, NE_LIFT_W1, NE_LIFT_G1, NE_LIFT_B1,
             NE_LIFT_W2, NE_LIFT_G2, NE_LIFT_B2, bufE2, En);             // edge1

        // ---- SplitLayer0_1 edge<->cycle ----
        MS(c1a, (size_t)C1n * Dn);
        SCAT(c1a, edge, e2ce, e2cn, P2n);                                // a
        MS(segC, (size_t)Cn * Dn);
        SCAT(segC, c1a, nullptr, cdom, C1n);
        k_gather<<<NB((long long)C1n * Dn), 256, 0, stream>>>(c1b, segC, cdom,
                (long long)C1n * Dn);                                    // dom_a
        GEMM(c1a, 128, c1b, 128, cyc, 128, 3, Lp + EC_LVL1_W, 128, Ybig, C1n);
        BN(c1c, Ybig, Lp + EC_LVL1_G, Lp + EC_LVL1_B, C1n, 128, 0);      // lvl_c
        MS(bufE1, (size_t)En * Dn);
        SCAT(bufE1, c1c, e2cn, e2ce, P2n);                               // g0
        MS(segC, (size_t)Cn * Dn);
        SCAT(segC, c1c, nullptr, cdom, C1n);
        k_gather<<<NB((long long)C1n * Dn), 256, 0, stream>>>(c1d, segC, cdom,
                (long long)C1n * Dn);                                    // dom
        MS(bufE3, (size_t)En * Dn);
        SCAT(bufE3, c1d, e2cn, e2ce, P2n);                               // g1
        k_comb3<<<NB((long long)En * Dn), 256, 0, stream>>>(bufE1, edge, bufE1,
                bufE3, Lp + EC_EPS1, Lp + EC_AFF, (long long)En * Dn);   // t3
        MLP2(bufE1, 128, nullptr, 0, 1, Lp, EC_LVL2_W1, EC_LVL2_G1, EC_LVL2_B1,
             EC_LVL2_W2, EC_LVL2_G2, EC_LVL2_B2, bufE3, En);             // edge2
        MS(cnt, (size_t)Cn);
        k_count<<<NB(C1n), 256, 0, stream>>>(cnt, cdom, C1n);
        MS(segC, (size_t)Cn * Dn);
        SCAT(segC, cyc, nullptr, cdom, C1n);
        k_pre2<<<NB((long long)C1n * Dn), 256, 0, stream>>>(c1a, c1b, cyc, segC,
                cnt, cdom, Lp + EC_EPS2, (long long)C1n * Dn);           // pre
        MLP2(c1a, 128, c1b, 128, 2, Lp, EC_LIFT_W1, EC_LIFT_G1, EC_LIFT_B1,
             EC_LIFT_W2, EC_LIFT_G2, EC_LIFT_B2, cycN, C1n);             // cyc_new
        GEMM(bufE2, 128, bufE3, 128, nullptr, 0, 2, Lp + MLP_W, 128, Ybig, En);
        BN(edgeN, Ybig, Lp + MLP_G, Lp + MLP_B, En, 128, 0);             // edge = mlp

        float* tp;
        tp = node; node = nodeN; nodeN = tp;
        tp = edge; edge = edgeN; edgeN = tp;
        tp = cyc;  cyc  = cycN;  cycN  = tp;
    }

    // ---- pooling + head ----
    MS(r0, (size_t)Bn * Dn); SCAT(r0, node, nullptr, batch, Nn);
    MS(r1, (size_t)Bn * Dn); SCAT(r1, edge, nullptr, ebatch, En);
    MS(r2, (size_t)Bn * Dn); SCAT(r2, cyc, nullptr, cbatch, C1n);
    MS(rep, (size_t)Bn * 256);
    float* rs[3] = { r0, r1, r2 };
    for (int j = 0; j < 3; ++j) {
        const float* pp = prm + OFF_POOL + (size_t)j * POOL_SZ;
        GEMM(rs[j], 128, nullptr, 0, nullptr, 0, 1, pp, 256, Ybig, Bn);
        BN(rep, Ybig, pp + 33024, pp + 32768, Bn, 256, 1);   // accumulate
    }
    k_final<<<NB(Bn), 256, 0, stream>>>(out, rep, prm + OFF_LINW, prm + OFF_LINB, Bn);
}
